// Mamba_30468497998114
// MI455X (gfx1250) — matrix operand order repaired and checked by reading, not compiled, not run on hardware
//
#include <hip/hip_runtime.h>
#include <hip/hip_bf16.h>
#include <math.h>

// ---------------- problem constants ----------------
#define BATCH     2
#define SEQL      1024
#define DMODEL    2048
#define DSTATE    64
#define HEADDIM   64
#define NHEADS    32                       // DMODEL/HEADDIM
#define DCONV     4
#define CONVDIM   (DMODEL + 2*DSTATE)      // 2176
#define NPROJ     (2*DMODEL + 2*DSTATE + NHEADS)  // 4256
#define NPROJ_PAD 4288                     // round up to multiple of 64
#define MROWS     (BATCH*SEQL)             // 2048

typedef __attribute__((ext_vector_type(16))) _Float16 v16h;
typedef __attribute__((ext_vector_type(8)))  _Float16 v8h;
typedef __attribute__((ext_vector_type(8)))  float    v8f;

union V16U { v16h v; v8h h[2]; };

// ---------------- WMMA fragment loads (wave32 layouts, ISA 7.12.2) ----------------
__device__ inline v16h load_frag_A(const _Float16* __restrict__ A, int lda,
                                   int mbase, int kb, int lane) {
    int row  = mbase + (lane & 15);
    int koff = (lane & 16) >> 1;              // 0 or 8
    const _Float16* p = A + (size_t)row * lda + kb + koff;
    V16U u;
    u.h[0] = *(const v8h*)(p);                // K = koff..koff+7
    u.h[1] = *(const v8h*)(p + 16);           // K = 16+koff..23+koff
    return u.v;
}

__device__ inline v16h load_frag_B(const _Float16* __restrict__ W, int ldb,
                                   int nb, int kb, int lane) {
    int col = nb + (lane & 15);
    int ks  = kb + ((lane & 16) >> 1);        // 0 or 8
    const _Float16* p = W + (size_t)col * ldb + ks;
    V16U u;
    u.h[0] = *(const v8h*)(p);                // K = ks..ks+7
    u.h[1] = *(const v8h*)(p + 16);           // K = 16+ks..23+ks
    return u.v;
}

// ---------------- f16 WMMA GEMM: C[M,Nact] = A[M,K] * W[N,K]^T ----------------
// one wave -> 32x64 tile: 2 A-frags x 4 B-frags = 8 wmma per K-step (step 32)
__global__ __launch_bounds__(128)
void mamba_gemm_wmma(const _Float16* __restrict__ A, const _Float16* __restrict__ W,
                     float* __restrict__ C, int M, int K, int Nact,
                     int lda, int ldb, int ldc, int ngroups) {
    int lane = threadIdx.x & 31;
    int wave = threadIdx.x >> 5;
    int gw   = blockIdx.x * (blockDim.x >> 5) + wave;
    int mt   = gw / ngroups;
    int ng   = gw % ngroups;
    if (mt * 32 >= M) return;
    int mbase = mt * 32;
    int nbase = ng * 64;

    v8f acc[2][4];
    #pragma unroll
    for (int i = 0; i < 2; ++i)
        #pragma unroll
        for (int t = 0; t < 4; ++t)
            acc[i][t] = (v8f){0.f,0.f,0.f,0.f,0.f,0.f,0.f,0.f};

    for (int kb = 0; kb < K; kb += 32) {
        v16h a0 = load_frag_A(A, lda, mbase,      kb, lane);
        v16h a1 = load_frag_A(A, lda, mbase + 16, kb, lane);
        #pragma unroll
        for (int t = 0; t < 4; ++t) {
            v16h b = load_frag_B(W, ldb, nbase + t * 16, kb, lane);
            acc[0][t] = __builtin_amdgcn_wmma_f32_16x16x32_f16(false, a0, false, b, (short)0, acc[0][t], false, false);
            acc[1][t] = __builtin_amdgcn_wmma_f32_16x16x32_f16(false, a1, false, b, (short)0, acc[1][t], false, false);
        }
    }

    int ncol0 = lane & 15;
    #pragma unroll
    for (int i = 0; i < 2; ++i) {
        int mrow = mbase + i * 16 + ((lane >> 4) << 3);   // +0 or +8
        #pragma unroll
        for (int t = 0; t < 4; ++t) {
            int ncol = nbase + t * 16 + ncol0;
            if (ncol < Nact) {
                #pragma unroll
                for (int r = 0; r < 8; ++r)
                    C[(size_t)(mrow + r) * ldc + ncol] = acc[i][t][r];
            }
        }
    }
}

// ---------------- converters ----------------
__global__ void k_f32_to_f16(const float* __restrict__ s, _Float16* __restrict__ d, size_t n) {
    size_t i = (size_t)blockIdx.x * blockDim.x + threadIdx.x;
    if (i < n) d[i] = (_Float16)s[i];
}

// W_in [rows_src, cols] -> f16 [rows_dst, cols], zero-pad extra rows
__global__ void k_convert_pad_rows(const float* __restrict__ s, _Float16* __restrict__ d,
                                   int rows_src, int rows_dst, int cols) {
    size_t i = (size_t)blockIdx.x * blockDim.x + threadIdx.x;
    size_t n = (size_t)rows_dst * cols;
    if (i >= n) return;
    int r = (int)(i / cols);
    int c = (int)(i % cols);
    d[i] = (r < rows_src) ? (_Float16)s[(size_t)r * cols + c] : (_Float16)0.f;
}

// ---------------- causal depthwise conv (K=4, left pad 3) + SiLU ----------------
__global__ void k_conv_silu(const float* __restrict__ proj, const float* __restrict__ cw,
                            const float* __restrict__ cb, float* __restrict__ xbc) {
    size_t i = (size_t)blockIdx.x * blockDim.x + threadIdx.x;
    size_t n = (size_t)MROWS * CONVDIM;
    if (i >= n) return;
    int row = (int)(i / CONVDIM);         // b*L + l
    int c   = (int)(i % CONVDIM);
    int l   = row & (SEQL - 1);
    float acc = cb[c];
    #pragma unroll
    for (int k = 0; k < DCONV; ++k) {
        int lt = l - (DCONV - 1) + k;
        if (lt >= 0)
            acc = fmaf(cw[c * DCONV + k],
                       proj[(size_t)(row - (DCONV - 1) + k) * NPROJ_PAD + DMODEL + c], acc);
    }
    xbc[i] = acc / (1.f + expf(-acc));    // SiLU
}

// ---------------- dt = softplus(dt_tok + bias); dA = exp(dt * -exp(A_log)) ----------------
__global__ void k_dt_dA(const float* __restrict__ proj, const float* __restrict__ dt_bias,
                        const float* __restrict__ A_log,
                        float* __restrict__ dtb, float* __restrict__ dAb) {
    size_t i = (size_t)blockIdx.x * blockDim.x + threadIdx.x;
    size_t n = (size_t)MROWS * NHEADS;
    if (i >= n) return;
    int row = (int)(i / NHEADS);
    int h   = (int)(i % NHEADS);
    float x  = proj[(size_t)row * NPROJ_PAD + (2 * DMODEL + 2 * DSTATE) + h] + dt_bias[h];
    float dt = (x > 20.f) ? x : log1pf(expf(x));
    float Ah = -expf(A_log[h]);
    dtb[i] = dt;
    dAb[i] = expf(dt * Ah);
}

// ---------------- sequential SSM scan: 1 workgroup per (b,h); state in registers ----------------
// thread t: p = t>>2, n-range = (t&3)*16 .. +16 ; 4-lane shfl reduction over N
__global__ __launch_bounds__(256)
void k_scan(const float* __restrict__ xbc, const float* __restrict__ dtb,
            const float* __restrict__ dAb, const float* __restrict__ Dp,
            float* __restrict__ y) {
    int bh = blockIdx.x;
    int b  = bh >> 5;
    int h  = bh & 31;
    int t  = threadIdx.x;
    int p  = t >> 2;
    int n0 = (t & 3) * 16;
    float S[16];
    #pragma unroll
    for (int j = 0; j < 16; ++j) S[j] = 0.f;
    float Dh = Dp[h];

    for (int l = 0; l < SEQL; ++l) {
        size_t row = (size_t)b * SEQL + l;
        float dAv = dAb[row * NHEADS + h];
        float dtv = dtb[row * NHEADS + h];
        const float* xr = xbc + row * CONVDIM;
        float xp   = xr[h * HEADDIM + p];
        float coef = dtv * xp;
        const float4* Bv = (const float4*)(xr + DMODEL + n0);
        const float4* Cv = (const float4*)(xr + DMODEL + DSTATE + n0);
        float acc = 0.f;
        #pragma unroll
        for (int j = 0; j < 4; ++j) {
            float4 Bj = Bv[j];
            float4 Cj = Cv[j];
            S[4*j+0] = fmaf(S[4*j+0], dAv, coef * Bj.x); acc = fmaf(S[4*j+0], Cj.x, acc);
            S[4*j+1] = fmaf(S[4*j+1], dAv, coef * Bj.y); acc = fmaf(S[4*j+1], Cj.y, acc);
            S[4*j+2] = fmaf(S[4*j+2], dAv, coef * Bj.z); acc = fmaf(S[4*j+2], Cj.z, acc);
            S[4*j+3] = fmaf(S[4*j+3], dAv, coef * Bj.w); acc = fmaf(S[4*j+3], Cj.w, acc);
        }
        acc += __shfl_xor(acc, 1, 32);
        acc += __shfl_xor(acc, 2, 32);
        if ((t & 3) == 0)
            y[row * DMODEL + h * HEADDIM + p] = acc + Dh * xp;
    }
}

// ---------------- gate: yg = y * silu(z), convert to f16 for out-proj GEMM ----------------
__global__ void k_gate(const float* __restrict__ y, const float* __restrict__ proj,
                       _Float16* __restrict__ yg) {
    size_t i = (size_t)blockIdx.x * blockDim.x + threadIdx.x;
    size_t n = (size_t)MROWS * DMODEL;
    if (i >= n) return;
    int row = (int)(i / DMODEL);
    int d   = (int)(i % DMODEL);
    float z = proj[(size_t)row * NPROJ_PAD + d];
    float g = z / (1.f + expf(-z));
    yg[i] = (_Float16)(y[i] * g);
}

// ---------------- workspace layout (all offsets multiple of 256B) ----------------
static constexpr size_t OFF_UA   = 0;                                          //  8 MB f16 [2048,2048]
static constexpr size_t OFF_WIN  = OFF_UA   + (size_t)MROWS*DMODEL*2;          // 17.6 MB f16 [4288,2048]
static constexpr size_t OFF_PROJ = OFF_WIN  + (size_t)NPROJ_PAD*DMODEL*2;      // 35 MB f32 [2048,4288]
static constexpr size_t OFF_XBC  = OFF_PROJ + (size_t)MROWS*NPROJ_PAD*4;       // 17.8 MB f32 [2048,2176]
static constexpr size_t OFF_DT   = OFF_XBC  + (size_t)MROWS*CONVDIM*4;         // 256 KB
static constexpr size_t OFF_DA   = OFF_DT   + (size_t)MROWS*NHEADS*4;          // 256 KB
static constexpr size_t OFF_Y    = OFF_DA   + (size_t)MROWS*NHEADS*4;          // 16.8 MB f32
static constexpr size_t OFF_YG   = OFF_Y    + (size_t)MROWS*DMODEL*4;          //  8 MB f16
static constexpr size_t OFF_WOUT = OFF_YG   + (size_t)MROWS*DMODEL*2;          //  8 MB f16
// total ~113 MB

extern "C" void kernel_launch(void* const* d_in, const int* in_sizes, int n_in,
                              void* d_out, int out_size, void* d_ws, size_t ws_size,
                              hipStream_t stream) {
    const float* u       = (const float*)d_in[0];
    const float* W_in    = (const float*)d_in[1];
    const float* conv_w  = (const float*)d_in[2];
    const float* conv_b  = (const float*)d_in[3];
    const float* W_out   = (const float*)d_in[4];
    const float* A_log   = (const float*)d_in[5];
    const float* D_param = (const float*)d_in[6];
    const float* dt_bias = (const float*)d_in[7];
    float* out = (float*)d_out;

    char* ws = (char*)d_ws;
    _Float16* uA   = (_Float16*)(ws + OFF_UA);
    _Float16* Win  = (_Float16*)(ws + OFF_WIN);
    float*    proj = (float*)   (ws + OFF_PROJ);
    float*    xbc  = (float*)   (ws + OFF_XBC);
    float*    dtb  = (float*)   (ws + OFF_DT);
    float*    dAb  = (float*)   (ws + OFF_DA);
    float*    yb   = (float*)   (ws + OFF_Y);
    _Float16* yg   = (_Float16*)(ws + OFF_YG);
    _Float16* Wout = (_Float16*)(ws + OFF_WOUT);

    const int TB = 256;

    // 1) precision converts
    {
        size_t n = (size_t)MROWS * DMODEL;                 // u -> f16
        k_f32_to_f16<<<(unsigned)((n + TB - 1) / TB), TB, 0, stream>>>(u, uA, n);
    }
    {
        size_t n = (size_t)NPROJ_PAD * DMODEL;             // W_in -> f16, zero-pad rows
        k_convert_pad_rows<<<(unsigned)((n + TB - 1) / TB), TB, 0, stream>>>(W_in, Win, NPROJ, NPROJ_PAD, DMODEL);
    }
    {
        size_t n = (size_t)DMODEL * DMODEL;                // W_out -> f16
        k_f32_to_f16<<<(unsigned)((n + TB - 1) / TB), TB, 0, stream>>>(W_out, Wout, n);
    }

    // 2) in-projection GEMM: proj[2048, 4256(pad 4288)] = uA * W_in^T
    {
        int ngroups = NPROJ_PAD / 64;                      // 67
        int waves   = (MROWS / 32) * ngroups;              // 64*67 = 4288
        int blocks  = waves / 4;                           // 128 thr = 4 waves
        mamba_gemm_wmma<<<blocks, 128, 0, stream>>>(uA, Win, proj, MROWS, DMODEL,
                                                    NPROJ, DMODEL, DMODEL, NPROJ_PAD, ngroups);
    }

    // 3) depthwise causal conv + SiLU
    {
        size_t n = (size_t)MROWS * CONVDIM;
        k_conv_silu<<<(unsigned)((n + TB - 1) / TB), TB, 0, stream>>>(proj, conv_w, conv_b, xbc);
    }

    // 4) dt / dA
    {
        size_t n = (size_t)MROWS * NHEADS;
        k_dt_dA<<<(unsigned)((n + TB - 1) / TB), TB, 0, stream>>>(proj, dt_bias, A_log, dtb, dAb);
    }

    // 5) sequential scan: 64 workgroups (b,h), state in registers
    k_scan<<<BATCH * NHEADS, 256, 0, stream>>>(xbc, dtb, dAb, D_param, yb);

    // 6) gate with silu(z), convert to f16
    {
        size_t n = (size_t)MROWS * DMODEL;
        k_gate<<<(unsigned)((n + TB - 1) / TB), TB, 0, stream>>>(yb, proj, yg);
    }

    // 7) out-projection GEMM: out[2048,2048] = yg * W_out^T
    {
        int ngroups = DMODEL / 64;                         // 32
        int waves   = (MROWS / 32) * ngroups;              // 64*32 = 2048
        int blocks  = waves / 4;
        mamba_gemm_wmma<<<blocks, 128, 0, stream>>>(yg, Wout, out, MROWS, DMODEL,
                                                    DMODEL, DMODEL, DMODEL, DMODEL, ngroups);
    }
}
